// MCShadingNetwork_64046552318094
// MI455X (gfx1250) — matrix operand order repaired and checked by reading, not compiled, not run on hardware
//
#include <hip/hip_runtime.h>
#include <hip/hip_bf16.h>
#include <math.h>

typedef _Float16 v16h __attribute__((ext_vector_type(16)));
typedef _Float16 v8h  __attribute__((ext_vector_type(8)));
typedef float    v8f  __attribute__((ext_vector_type(8)));

#define NPTS 512
#define ND 512
#define NS 256
#define NDIR_D (NPTS*ND)          // 262144
#define NDIR_S (NPTS*NS)          // 131072
#define TILES_D (NDIR_D/16)       // 16384
#define TILES_S (NDIR_S/16)       // 8192
#define TILES_TOT (TILES_D+TILES_S) // 24576

// ---------------- static device arenas ----------------
constexpr int OF_PE    = 0;                    // 512 x 51
constexpr int OF_CAT   = OF_PE    + NPTS*51;   // 512 x 307
constexpr int OF_FX    = OF_CAT   + NPTS*307;  // 512 x 259
constexpr int OF_T0    = OF_FX    + NPTS*259;  // 512 x 256
constexpr int OF_T1    = OF_T0    + NPTS*256;  // 512 x 256
constexpr int OF_FEATS = OF_T1    + NPTS*256;  // 512 x 256
constexpr int OF_METAL = OF_FEATS + NPTS*256;  // 512
constexpr int OF_ROUGH = OF_METAL + NPTS;      // 512
constexpr int OF_ALB   = OF_ROUGH + NPTS;      // 512 x 3
constexpr int FBUF_SZ  = OF_ALB   + NPTS*3;

__device__ float     g_fbuf[FBUF_SZ];
__device__ _Float16  g_packW[1310720];   // packed f16 weights (WMMA B layout)
__device__ float     g_packB[6144];      // padded biases
__device__ float     g_mat[17*36];       // IDE_MAT
__device__ float     g_nrm[NPTS*3], g_view[NPTS*3], g_refl[NPTS*3];
__device__ float     g_NoV[NPTS], g_kk[NPTS], g_F0[NPTS*3];
__device__ float     g_partDiff[TILES_D*3];
__device__ float     g_partSpec[TILES_S*3];

__constant__ int c_mIdx[36] = {
  0,1, 0,1,2, 0,1,2,3,4, 0,1,2,3,4,5,6,7,8,
  0,1,2,3,4,5,6,7,8,9,10,11,12,13,14,15,16 };

// ---------------- helpers ----------------
__device__ __forceinline__ v16h cat16(v8h lo, v8h hi){
  return __builtin_shufflevector(lo,hi,0,1,2,3,4,5,6,7,8,9,10,11,12,13,14,15);
}

__device__ double d_fact(int n){ double r=1.0; for (int i=2;i<=n;++i) r*=(double)i; return r; }

__device__ __forceinline__ void orthoFrame(float zx,float zy,float zz,
                                           float& xx,float& xy,float& xz,
                                           float& yx,float& yy,float& yz){
  float o0x= zy, o0y=-zx, o0z=0.f;
  float o1x=-zz, o1y=0.f, o1z=zx;
  float n0 = o0x*o0x+o0y*o0y;
  float n1 = o1x*o1x+o1z*o1z;
  float ox,oy,oz;
  if (n0>n1){ox=o0x;oy=o0y;oz=o0z;} else {ox=o1x;oy=o1y;oz=o1z;}
  float inv = rsqrtf(ox*ox+oy*oy+oz*oz+1e-8f);
  xx=ox*inv; xy=oy*inv; xz=oz*inv;
  yx = zy*xz - zz*xy;  yy = zz*xx - zx*xz;  yz = zx*xy - zy*xx;
}

// ---------------- init: IDE coefficient matrix (17 x 36) ----------------
__global__ void k_init_mat(){
  if (threadIdx.x != 0 || blockIdx.x != 0) return;
  for (int i=0;i<17*36;++i) g_mat[i]=0.f;
  int idx=0;
  for (int li=0; li<5; ++li){
    int l = 1<<li;
    double p2 = 1.0; for (int q=0;q<l;++q) p2 *= 2.0;
    for (int m=0;m<=l;++m){
      for (int k=0;k<=l-m;++k){
        double a = 0.5*((double)l + (double)k + (double)m - 1.0);
        double gb = 1.0;
        for (int q=0;q<l;++q) gb *= (a - (double)q);
        gb /= d_fact(l);
        double assoc = ((m&1)?-1.0:1.0) * p2 * d_fact(l)/d_fact(k)/d_fact(l-k-m)*gb;
        double coeff = sqrt((2.0*(double)l+1.0)*d_fact(l-m)/(4.0*3.14159265358979323846*d_fact(l+m)))*assoc;
        g_mat[k*36+idx] = (float)coeff;
      }
      ++idx;
    }
  }
}

// ---------------- per-point prep: normalize, refl, NoV, pos_enc ----------------
__global__ void k_prep(const float* pts, const float* normals, const float* view){
  int p = blockIdx.x*blockDim.x + threadIdx.x;
  if (p >= NPTS) return;
  float nx=normals[p*3], ny=normals[p*3+1], nz=normals[p*3+2];
  float inv = rsqrtf(nx*nx+ny*ny+nz*nz+1e-8f); nx*=inv; ny*=inv; nz*=inv;
  g_nrm[p*3]=nx; g_nrm[p*3+1]=ny; g_nrm[p*3+2]=nz;
  float vx=view[p*3], vy=view[p*3+1], vz=view[p*3+2];
  inv = rsqrtf(vx*vx+vy*vy+vz*vz+1e-8f); vx*=inv; vy*=inv; vz*=inv;
  g_view[p*3]=vx; g_view[p*3+1]=vy; g_view[p*3+2]=vz;
  float dd = nx*vx+ny*vy+nz*vz;
  g_NoV[p] = fminf(fmaxf(dd,0.f),1.f);
  g_refl[p*3]   = 2.f*dd*nx - vx;
  g_refl[p*3+1] = 2.f*dd*ny - vy;
  g_refl[p*3+2] = 2.f*dd*nz - vz;
  float px=pts[p*3], py=pts[p*3+1], pz=pts[p*3+2];
  g_fbuf[OF_PE+p*51+0]=px; g_fbuf[OF_PE+p*51+1]=py; g_fbuf[OF_PE+p*51+2]=pz;
  for (int f=0; f<8; ++f){
    float fr = (float)(1<<f);
    float a0=px*fr, a1=py*fr, a2=pz*fr;
    int base = OF_PE + p*51 + 3 + f*6;
    g_fbuf[base+0]=sinf(a0); g_fbuf[base+1]=sinf(a1); g_fbuf[base+2]=sinf(a2);
    g_fbuf[base+3]=cosf(a0); g_fbuf[base+4]=cosf(a1); g_fbuf[base+5]=cosf(a2);
  }
}

// ---------------- repack f32 weights into WMMA B-operand f16 layout ----------------
__global__ void k_repack(const float* W, int K, int N, int Kp, int Np, int dstOff){
  int tid = blockIdx.x*blockDim.x + threadIdx.x;
  int ntiles = Np >> 4;
  int total = (Kp>>5)*ntiles*32;
  if (tid >= total) return;
  int lane = tid & 31, tt = tid >> 5;
  int kt = tt / ntiles, nt = tt - kt*ntiles;
  int col = nt*16 + (lane & 15);
  int kBase = kt*32 + ((lane & 16) ? 16 : 0);
  _Float16* dst = g_packW + dstOff + (size_t)tt*512 + lane*16;
  for (int e=0;e<16;++e){
    int kk = kBase + e;
    float v = (kk < K && col < N) ? W[(size_t)kk*N + col] : 0.f;
    dst[e] = (_Float16)v;
  }
}

__global__ void k_padbias(const float* b, int N, int Np, int dstOff){
  int t = threadIdx.x;
  if (t < Np) g_packB[dstOff+t] = (t < N) ? b[t] : 0.f;
}

// ---------------- core WMMA tile loop: A held in registers ----------------
// aReg: KT fragments of the wave's 16-row A block.  For each output column
// tile: stream packed-B with immediate offsets, one WMMA per kt.
template<int KT>
__device__ __forceinline__ void load_a_frags(const _Float16* aIn, int Kp, v16h* aReg){
  const int lane = threadIdx.x & 31;
  const int l15 = lane & 15, hl = lane >> 4;
#pragma unroll
  for (int kt=0; kt<KT; ++kt){
    const _Float16* ap = aIn + l15*Kp + kt*32 + hl*16;
    aReg[kt] = cat16(*(const v8h*)ap, *(const v8h*)(ap+8));
  }
}

template<int KT>
__device__ __forceinline__ v8f wmma_col(const v16h* aReg, const _Float16* Wp,
                                        int nt, int ntiles, float bv){
  const int lane = threadIdx.x & 31;
  v8f acc = {bv,bv,bv,bv,bv,bv,bv,bv};
  const _Float16* wb = Wp + (size_t)nt*512 + lane*16;
#pragma unroll
  for (int kt=0; kt<KT; ++kt){
    const _Float16* wp = wb + (size_t)kt*ntiles*512;
    v8h b0 = *(const v8h*)wp;
    v8h b1 = *(const v8h*)(wp+8);
    acc = __builtin_amdgcn_wmma_f32_16x16x32_f16(false, aReg[kt],
                                                 false, cat16(b0,b1),
                                                 (short)0, acc, false, false);
  }
  return acc;
}

// ---------------- wave-local fused MLP layer (Np=256, out stride 256) ----------
template<int KT>
__device__ __forceinline__ void wave_layer(const _Float16* aIn, _Float16* aOut,
                                           const _Float16* Wp, const float* bp)
{
  const int lane = threadIdx.x & 31;
  const int l15 = lane & 15, hl = lane >> 4;
  v16h aReg[KT];
  load_a_frags<KT>(aIn, KT*32, aReg);
  for (int nt=0; nt<16; ++nt){
    if (nt+1 < 16)
      __builtin_prefetch(Wp + (size_t)(nt+1)*512 + lane*16, 0, 1);
    v8f acc = wmma_col<KT>(aReg, Wp, nt, 16, bp[nt*16+l15]);
#pragma unroll
    for (int v=0;v<8;++v){
      float x = fmaxf(acc[v],0.f);   // relu
      aOut[(hl*8+v)*256 + nt*16 + l15] = (_Float16)x;
    }
  }
}

// ---------------- generic WMMA GEMM: C = act(A @ W + b), M=512 ----------------
template<int KT>
__global__ void k_gemm(int aOff, int K,
                       int wOff, int bOff, int cOff, int N, int Np, int act)
{
  extern __shared__ _Float16 sA[];
  const int Kp = KT*32;
  const int wid  = threadIdx.x >> 5;
  const int lane = threadIdx.x & 31;
  _Float16* aBuf = sA + wid * (Kp*16);
  const int r0 = (blockIdx.x*(blockDim.x>>5) + wid)*16;
  const float* A = g_fbuf + aOff;
  for (int r=0;r<16;++r){
    const float* src = A + (size_t)(r0+r)*K;
    for (int k=lane; k<Kp; k+=32)
      aBuf[r*Kp+k] = (k<K) ? (_Float16)src[k] : (_Float16)0.f;
  }
  __syncthreads();
  const _Float16* Wp = g_packW + wOff;
  const float* bp = g_packB + bOff;
  float* C = g_fbuf + cOff;
  const int l15 = lane & 15, hl = lane >> 4;
  const int ntiles = Np>>4;
  v16h aReg[KT];
  load_a_frags<KT>(aBuf, Kp, aReg);
  for (int nt=0; nt<ntiles; ++nt){
    v8f acc = wmma_col<KT>(aReg, Wp, nt, ntiles, bp[nt*16+l15]);
#pragma unroll
    for (int v=0; v<8; ++v){
      float x = acc[v];
      if (act==0)      x = fmaxf(x,0.f);
      else if (act==2) x = 1.f/(1.f+__expf(-x));
      int row = r0 + hl*8 + v;
      int col = nt*16 + l15;
      if (col < N && row < NPTS) C[(size_t)row*N + col] = x;
    }
  }
}

// ---------------- fused outer_light + reductions ----------------
#define OUTER_WAVE_BYTES (2*16*256*2 + 16*4*4)  // two 16x256 f16 act buffers + 16x4 dirs

__global__ void k_outer(const float* dsamp, const float* ssamp,
                        int w1,int w2,int w3,int w4,
                        int b1,int b2,int b3,int b4)
{
  extern __shared__ char sraw[];
  const int wid  = threadIdx.x >> 5;
  const int lane = threadIdx.x & 31;
  char* wbase = sraw + wid*OUTER_WAVE_BYTES;
  _Float16* buf0 = (_Float16*)wbase;
  _Float16* buf1 = buf0 + 16*256;
  float* dirsL = (float*)(wbase + 2*16*256*2);
  const int t = blockIdx.x*(blockDim.x>>5) + wid;
  const int r = lane & 15;
  const int g = t*16 + r;
  const bool isDiff = (t < TILES_D);

  int p; float dx,dy,dz;
  if (isDiff){
    p = g >> 9; int s = g & 511;
    float zx=g_nrm[p*3], zy=g_nrm[p*3+1], zz=g_nrm[p*3+2];
    float xx,xy,xz,yx,yy,yz; orthoFrame(zx,zy,zz,xx,xy,xz,yx,yy,yz);
    float az = dsamp[s*2]*6.28318530717958647692f;
    float el = dsamp[s*2+1];
    float es = sqrtf(el + 1e-7f);
    float cx = es*cosf(az), cy = es*sinf(az), cz = sqrtf(1.f-el+1e-7f);
    dx = cx*xx+cy*yx+cz*zx; dy = cx*xy+cy*yy+cz*zy; dz = cx*xz+cy*yz+cz*zz;
  } else {
    int g2 = g - NDIR_D; p = g2 >> 8; int s = g2 & 255;
    float zx=g_refl[p*3], zy=g_refl[p*3+1], zz=g_refl[p*3+2];
    float xx,xy,xz,yx,yy,yz; orthoFrame(zx,zy,zz,xx,xy,xz,yx,yy,yz);
    float phi = 6.28318530717958647692f*ssamp[s*2];
    float el  = ssamp[s*2+1];
    float rough = g_fbuf[OF_ROUGH + p];
    float a2 = rough*rough;
    float ct = sqrtf((1.f-el+1e-6f)/(1.f+(a2-1.f)*el+1e-6f) + 1e-6f);
    float st = sqrtf(fmaxf(1.f-ct*ct, 1e-6f));
    float cx = st*cosf(phi), cy = st*sinf(phi);
    dx = cx*xx+cy*yx+ct*zx; dy = cx*xy+cy*yy+ct*zy; dz = cx*xz+cy*yz+ct*zz;
    float inv = rsqrtf(dx*dx+dy*dy+dz*dz+1e-8f); dx*=inv; dy*=inv; dz*=inv;
  }
  if (lane < 16){ dirsL[r*4]=dx; dirsL[r*4+1]=dy; dirsL[r*4+2]=dz; }

  // IDE features (72 dims, pad to 96) into buf0 (row stride 96)
  float zp[17]; zp[0]=1.f;
#pragma unroll
  for (int k=1;k<17;++k) zp[k]=zp[k-1]*dz;
  float crr[17], cii[17]; crr[0]=1.f; cii[0]=0.f;
#pragma unroll
  for (int m=1;m<17;++m){ crr[m]=crr[m-1]*dx - cii[m-1]*dy; cii[m]=crr[m-1]*dy + cii[m-1]*dx; }
  const int jb = (lane>>4)*18;
  for (int jj=0;jj<18;++jj){
    int j = jb+jj;
    float vz = 0.f;
    for (int k=0;k<17;++k) vz += zp[k]*g_mat[k*36+j];
    int m = c_mIdx[j];
    buf0[r*96 + j]      = (_Float16)(crr[m]*vz);
    buf0[r*96 + 36 + j] = (_Float16)(cii[m]*vz);
  }
  {
    int base = 72 + (lane>>4)*12;
    for (int q=0;q<12;++q) buf0[r*96+base+q] = (_Float16)0.f;
  }
  __syncthreads();
  wave_layer<3>(buf0, buf1, g_packW+w1, g_packB+b1);
  __syncthreads();
  wave_layer<8>(buf1, buf0, g_packW+w2, g_packB+b2);
  __syncthreads();
  wave_layer<8>(buf0, buf1, g_packW+w3, g_packB+b3);
  __syncthreads();

  // final layer: 256 -> 16 (cols 0..2 valid) kept in registers
  const int l15 = lane & 15, hl = lane >> 4;
  v16h aReg[8];
  load_a_frags<8>(buf1, 256, aReg);
  v8f acc = wmma_col<8>(aReg, g_packW+w4, 0, 1, g_packB[b4 + l15]);
  float L[8];
#pragma unroll
  for (int v=0;v<8;++v) L[v] = __expf(fminf(acc[v], 5.f));

  float ssum = 0.f;
  if (isDiff){
#pragma unroll
    for (int v=0;v<8;++v) ssum += L[v];
  } else {
    int c = (l15 < 3) ? l15 : 0;
    float vwx=g_view[p*3], vwy=g_view[p*3+1], vwz=g_view[p*3+2];
    float nx=g_nrm[p*3],  ny=g_nrm[p*3+1],  nz=g_nrm[p*3+2];
    float nov=g_NoV[p], kk=g_kk[p], f0=g_F0[p*3+c];
    float ga = nov/(nov*(1.f-kk)+kk+1e-7f);
#pragma unroll
    for (int v=0;v<8;++v){
      int rr = hl*8+v;
      float ddx=dirsL[rr*4], ddy=dirsL[rr*4+1], ddz=dirsL[rr*4+2];
      float hx=vwx+ddx, hy=vwy+ddy, hz=vwz+ddz;
      float hn = rsqrtf(hx*hx+hy*hy+hz*hz+1e-8f);
      float hov = (hx*vwx+hy*vwy+hz*vwz)*hn;
      hov = fminf(fmaxf(hov,0.f),1.f);
      float omh = 1.f-hov; float o2=omh*omh;
      float fr = f0 + (1.f-f0)*o2*o2*omh;
      float nol = nx*ddx+ny*ddy+nz*ddz;
      nol = fminf(fmaxf(nol,0.f),1.f);
      float geo = ga * (nol/(nol*(1.f-kk)+kk+1e-7f));
      ssum += fr*geo*L[v];
    }
  }
  ssum += __shfl_xor(ssum, 16);
  if (lane < 3){
    if (isDiff) g_partDiff[t*3+lane] = ssum;
    else        g_partSpec[(t-TILES_D)*3+lane] = ssum;
  }
}

// ---------------- small elementwise kernels ----------------
__global__ void k_concat(int aOff, int na, const float* Bext, int bOff, int nb, int dstOff){
  int idx = blockIdx.x*blockDim.x+threadIdx.x;
  int w = na+nb;
  if (idx >= NPTS*w) return;
  int rr = idx / w, c = idx - rr*w;
  const float* B = Bext ? Bext : (g_fbuf + bOff);
  float v = (c < na) ? g_fbuf[aOff + rr*na + c] : B[rr*nb + (c-na)];
  g_fbuf[dstOff + idx] = v;
}

__global__ void k_f0(){
  int p = blockIdx.x*blockDim.x+threadIdx.x;
  if (p>=NPTS) return;
  float m  = g_fbuf[OF_METAL+p];
  float rg = g_fbuf[OF_ROUGH+p];
  g_kk[p] = (rg+1.f)*(rg+1.f)*0.125f;
  for (int c=0;c<3;++c)
    g_F0[p*3+c] = 0.04f*(1.f-m) + g_fbuf[OF_ALB+p*3+c]*m;
}

__global__ void k_combine(float* out){
  int tid = blockIdx.x*blockDim.x+threadIdx.x;
  if (tid >= NPTS*3) return;
  int p = tid/3, c = tid - p*3;
  float d=0.f;
  for (int i=0;i<32;++i) d += g_partDiff[(p*32+i)*3+c];
  float s=0.f;
  for (int i=0;i<16;++i) s += g_partSpec[(p*16+i)*3+c];
  out[tid] = g_fbuf[OF_ALB+p*3+c]*(d*(1.f/512.f)) + s*(1.f/256.f);
}

// ---------------- host orchestration ----------------
extern "C" void kernel_launch(void* const* d_in, const int* in_sizes, int n_in,
                              void* d_out, int out_size, void* d_ws, size_t ws_size,
                              hipStream_t stream)
{
  (void)in_sizes; (void)n_in; (void)d_ws; (void)ws_size; (void)out_size;
  const float* pts     = (const float*)d_in[0];
  const float* normals = (const float*)d_in[1];
  const float* view    = (const float*)d_in[2];
  const float* dsamp   = (const float*)d_in[3];
  const float* ssamp   = (const float*)d_in[4];
  const float* P[48];
  for (int i=0;i<48;++i) P[i] = (const float*)d_in[5+i];

  k_init_mat<<<1,1,0,stream>>>();
  k_prep<<<(NPTS+255)/256,256,0,stream>>>(pts, normals, view);

  int wOff = 0, bOff = 0;
  auto packW = [&](const float* W,int K,int N,int Kp,int Np)->int{
    int off = wOff; wOff += (Kp/32)*(Np/16)*512;
    int threads = (Kp/32)*(Np/16)*32;
    k_repack<<<(threads+255)/256,256,0,stream>>>(W,K,N,Kp,Np,off);
    return off;
  };
  auto packB = [&](const float* b,int N,int Np)->int{
    int off = bOff; bOff += Np;
    k_padbias<<<1,Np,0,stream>>>(b,N,Np,off);
    return off;
  };
  auto gemm = [&](int aOff,int K,int Kp,int wo,int bo,int cOff,int N,int Np,int act){
    size_t smem = (size_t)8*Kp*16*2;
    switch (Kp>>5){
      case 2:  k_gemm<2> <<<NPTS/128,256,smem,stream>>>(aOff,K,wo,bo,cOff,N,Np,act); break;
      case 8:  k_gemm<8> <<<NPTS/128,256,smem,stream>>>(aOff,K,wo,bo,cOff,N,Np,act); break;
      case 9:  k_gemm<9> <<<NPTS/128,256,smem,stream>>>(aOff,K,wo,bo,cOff,N,Np,act); break;
      case 10: k_gemm<10><<<NPTS/128,256,smem,stream>>>(aOff,K,wo,bo,cOff,N,Np,act); break;
      default: break;
    }
  };

  // ---- pack point-pipeline weights ----
  int wf0[4], bf0[4], wf1[4], bf1[4];
  wf0[0]=packW(P[0],51,256,64,256);    bf0[0]=packB(P[1],256,256);
  wf0[1]=packW(P[2],256,256,256,256);  bf0[1]=packB(P[3],256,256);
  wf0[2]=packW(P[4],256,256,256,256);  bf0[2]=packB(P[5],256,256);
  wf0[3]=packW(P[6],256,256,256,256);  bf0[3]=packB(P[7],256,256);
  wf1[0]=packW(P[8],307,256,320,256);  bf1[0]=packB(P[9],256,256);
  wf1[1]=packW(P[10],256,256,256,256); bf1[1]=packB(P[11],256,256);
  wf1[2]=packW(P[12],256,256,256,256); bf1[2]=packB(P[13],256,256);
  wf1[3]=packW(P[14],256,256,256,256); bf1[3]=packB(P[15],256,256);

  // ---- feats0 (all relu) ----
  gemm(OF_PE, 51, 64,  wf0[0], bf0[0], OF_T0, 256,256, 0);
  gemm(OF_T0, 256,256, wf0[1], bf0[1], OF_T1, 256,256, 0);
  gemm(OF_T1, 256,256, wf0[2], bf0[2], OF_T0, 256,256, 0);
  gemm(OF_T0, 256,256, wf0[3], bf0[3], OF_T1, 256,256, 0);
  // cat307 = [h | pe]
  k_concat<<<(NPTS*307+255)/256,256,0,stream>>>(OF_T1,256, nullptr, OF_PE, 51, OF_CAT);
  // ---- feats1 (relu,relu,relu,linear) ----
  gemm(OF_CAT, 307,320, wf1[0], bf1[0], OF_T0, 256,256, 0);
  gemm(OF_T0,  256,256, wf1[1], bf1[1], OF_T1, 256,256, 0);
  gemm(OF_T1,  256,256, wf1[2], bf1[2], OF_T0, 256,256, 0);
  gemm(OF_T0,  256,256, wf1[3], bf1[3], OF_FEATS, 256,256, 1);
  // fx = [feats | pts]
  k_concat<<<(NPTS*259+255)/256,256,0,stream>>>(OF_FEATS,256, pts, 0, 3, OF_FX);

  // ---- heads (relu x3, sigmoid head) ----
  auto runHead = [&](int pi, int outOff, int Nout){
    int w0=packW(P[pi],259,256,288,256),    b0=packB(P[pi+1],256,256);
    int w1=packW(P[pi+2],256,256,256,256),  b1=packB(P[pi+3],256,256);
    int w2=packW(P[pi+4],256,256,256,256),  b2=packB(P[pi+5],256,256);
    int w3=packW(P[pi+6],256,Nout,256,16),  b3=packB(P[pi+7],Nout,16);
    gemm(OF_FX, 259,288, w0,b0, OF_T0, 256,256, 0);
    gemm(OF_T0, 256,256, w1,b1, OF_T1, 256,256, 0);
    gemm(OF_T1, 256,256, w2,b2, OF_T0, 256,256, 0);
    gemm(OF_T0, 256,256, w3,b3, outOff, Nout,16, 2);
  };
  runHead(16, OF_METAL, 1);
  runHead(24, OF_ROUGH, 1);
  runHead(32, OF_ALB,   3);
  k_f0<<<(NPTS+255)/256,256,0,stream>>>();

  // ---- outer light MLP (the hot path) ----
  int wo0=packW(P[40],72,256,96,256),    bo0=packB(P[41],256,256);
  int wo1=packW(P[42],256,256,256,256),  bo1=packB(P[43],256,256);
  int wo2=packW(P[44],256,256,256,256),  bo2=packB(P[45],256,256);
  int wo3=packW(P[46],256,3,256,16),     bo3=packB(P[47],3,16);

  size_t osmem = (size_t)8*OUTER_WAVE_BYTES;
  k_outer<<<TILES_TOT/8, 256, osmem, stream>>>(dsamp, ssamp,
                                               wo0,wo1,wo2,wo3,
                                               bo0,bo1,bo2,bo3);

  k_combine<<<(NPTS*3+255)/256,256,0,stream>>>((float*)d_out);
}
